// MultiHeadAttention_34351148434237
// MI455X (gfx1250) — matrix operand order repaired and checked by reading, not compiled, not run on hardware
//
#include <hip/hip_runtime.h>
#include <hip/hip_bf16.h>

// ---------------------------------------------------------------------------
// MultiHeadAttention for MI455X (gfx1250, wave32, WMMA).
// B=2, S=4096, D=512, H=8, hd=64.  All GEMM work on v_wmma_f32_16x16x32_f16.
// Attention inner loop: 64-key tiles -> 16 WMMA per softmax pass.
// ---------------------------------------------------------------------------

#define BB   2
#define SS   4096
#define DD   512
#define HH   8
#define HD   64
#define SCALE 0.125f   // 1/sqrt(64)

typedef _Float16 half4  __attribute__((ext_vector_type(4)));
typedef _Float16 half8  __attribute__((ext_vector_type(8)));
typedef _Float16 half16 __attribute__((ext_vector_type(16)));
typedef float    f32x4  __attribute__((ext_vector_type(4)));
typedef float    float8 __attribute__((ext_vector_type(8)));

__device__ __forceinline__ half16 h16cat(half8 lo, half8 hi) {
  return __builtin_shufflevector(lo, hi, 0,1,2,3,4,5,6,7,8,9,10,11,12,13,14,15);
}
__device__ __forceinline__ half8 cvt_h8(f32x4 a, f32x4 b) {
  half4 x = __builtin_convertvector(a, half4);
  half4 y = __builtin_convertvector(b, half4);
  return __builtin_shufflevector(x, y, 0,1,2,3,4,5,6,7);
}

// A fragment (16 x 32 f16): lane -> row = lane&15 ; K chunk split 8/8 per half-wave,
// vgprs 0..3 hold K = k0..k0+7, vgprs 4..7 hold K = 16+k0..16+k0+7 (k0 = 8*(lane>=16)).
__device__ __forceinline__ half16 load_frag_a(const _Float16* base, int lda,
                                              int row0, int kcol, int lane) {
  int m  = lane & 15;
  int k0 = (lane >> 4) * 8;
  const _Float16* p = base + (size_t)(row0 + m) * lda + kcol + k0;
  half8 lo = *(const half8*)p;        // K = k0 .. k0+7
  half8 hi = *(const half8*)(p + 16); // K = 16+k0 .. 16+k0+7
  return h16cat(lo, hi);
}

// A fragment sourced from f32 memory (converted to f16 in registers).
__device__ __forceinline__ half16 load_frag_a_f32(const float* base, int lda,
                                                  int row0, int kcol, int lane) {
  int m  = lane & 15;
  int k0 = (lane >> 4) * 8;
  const float* p = base + (size_t)(row0 + m) * lda + kcol + k0;
  f32x4 a0 = *(const f32x4*)p;
  f32x4 a1 = *(const f32x4*)(p + 4);
  f32x4 a2 = *(const f32x4*)(p + 16);
  f32x4 a3 = *(const f32x4*)(p + 20);
  return h16cat(cvt_h8(a0, a1), cvt_h8(a2, a3));
}

// B fragment (32 x 16 f16), supplied as B^T row-major [N][ldb]:
// lane -> col = lane&15 ; K split 8/8 per half-wave like A, k0 = 8*(lane>=16).
__device__ __forceinline__ half16 load_frag_b(const _Float16* bT, int ldb,
                                              int n0, int kcol, int lane) {
  int n  = lane & 15;
  int k0 = (lane >> 4) * 8;
  const _Float16* p = bT + (size_t)(n0 + n) * ldb + kcol + k0;
  half8 lo = *(const half8*)p;       // K = k0 .. k0+7
  half8 hi = *(const half8*)(p + 16); // K = 16+k0 .. 16+k0+7
  return h16cat(lo, hi);
}

__device__ __forceinline__ float8 wmma_f16(half16 a, half16 b, float8 c) {
  return __builtin_amdgcn_wmma_f32_16x16x32_f16(false, a, false, b,
                                                (short)0, c, false, false);
}

// ---------------------------------------------------------------------------
// Kernel 1: QKV projection.  Y = X @ W + b, f32 in, f16 out.
//   mode 0: out[((b*H+h)*S + s)*64 + f]   (Q, K: head-major row layout)
//   mode 1: out[((b*H+h)*64 + f)*S + s]   (V: transposed so PV B-frags are contiguous)
// Block: 256 thr (8 waves). Block tile: 128 rows x 64 cols; wave tile 16x64.
// ---------------------------------------------------------------------------
__global__ __launch_bounds__(256) void mha_proj_qkv(const float* __restrict__ X,
                                                    const float* __restrict__ W,
                                                    const float* __restrict__ bias,
                                                    _Float16* __restrict__ out,
                                                    int transposed) {
  __shared__ __align__(16) _Float16 wt[64 * 32];  // W^T tile: [64 cols][32 K]
  const int lane    = threadIdx.x & 31;
  const int wv      = threadIdx.x >> 5;
  const int rowbase = blockIdx.x * 128 + wv * 16;
  const int n0      = blockIdx.y * 64;
  const int hlf     = lane >> 4;
  const int nlane   = lane & 15;

  float8 acc[4] = {};

  for (int kk = 0; kk < DD; kk += 32) {
    __syncthreads();
    #pragma unroll
    for (int i = 0; i < 8; ++i) {            // stage W^T (f32 -> f16), 2048 elems
      int e   = threadIdx.x + i * 256;
      int col = e & 63;
      int kr  = e >> 6;
      wt[col * 32 + kr] = (_Float16)W[(size_t)(kk + kr) * DD + n0 + col];
    }
    __syncthreads();

    half16 a = load_frag_a_f32(X, DD, rowbase, kk, lane);
    #pragma unroll
    for (int c = 0; c < 4; ++c)
      acc[c] = wmma_f16(a, load_frag_b(wt, 32, c * 16, 0, lane), acc[c]);
  }

  #pragma unroll
  for (int c = 0; c < 4; ++c) {
    #pragma unroll
    for (int r = 0; r < 8; ++r) {
      int grow = rowbase + r + 8 * hlf;       // 0..8191
      int b    = grow >> 12;
      int s    = grow & (SS - 1);
      int col  = n0 + c * 16 + nlane;         // 0..511
      int h    = col >> 6;
      int f    = col & (HD - 1);
      float v  = acc[c][r] + bias[col];
      if (!transposed)
        out[(((size_t)(b * HH + h)) * SS + s) * HD + f] = (_Float16)v;
      else
        out[(((size_t)(b * HH + h)) * HD + f) * SS + s] = (_Float16)v;
    }
  }
}

// ---------------------------------------------------------------------------
// Kernel 2: flash attention.  Q,K: [B][H][S][64] f16, Vt: [B][H][64][S] f16.
// Block 256 thr = 8 waves, each wave owns 16 query rows x hd=64.
// Per 64-key tile: 8 WMMA (scores) + one online-softmax pass + 8 WMMA (PV).
// ---------------------------------------------------------------------------
__global__ __launch_bounds__(256) void mha_attn(const _Float16* __restrict__ Q,
                                                const _Float16* __restrict__ K,
                                                const _Float16* __restrict__ Vt,
                                                const unsigned char* __restrict__ mask,
                                                _Float16* __restrict__ O) {
  __shared__ __align__(16) _Float16 plds[8][16 * 64];  // per-wave P tile [16][64]
  const int lane  = threadIdx.x & 31;
  const int wv    = threadIdx.x >> 5;
  const int qbase = blockIdx.x * 128 + wv * 16;
  const int h     = blockIdx.y;
  const int b     = blockIdx.z;
  const int hlf   = lane >> 4;
  const int nlane = lane & 15;

  const _Float16* Qp = Q  + ((size_t)(b * HH + h)) * SS * HD;
  const _Float16* Kp = K  + ((size_t)(b * HH + h)) * SS * HD;
  const _Float16* Vp = Vt + ((size_t)(b * HH + h)) * HD * SS;
  const unsigned char* mp = mask + (size_t)b * SS * SS;

  // Q fragments resident for whole pass (hd = 64 -> two K=32 fragments).
  half16 qa0 = load_frag_a(Qp, HD, qbase, 0, lane);
  half16 qa1 = load_frag_a(Qp, HD, qbase, 32, lane);

  float8 o[4] = {};
  float mrow[8], lrow[8];
  #pragma unroll
  for (int r = 0; r < 8; ++r) { mrow[r] = -1e30f; lrow[r] = 0.0f; }

  for (int kb = 0; kb < SS; kb += 64) {
    if (kb + 64 < SS) {                       // global_prefetch_b8 next tiles
      __builtin_prefetch(Kp + (size_t)(kb + 64) * HD, 0, 1);
      __builtin_prefetch(Vp + (kb + 64), 0, 1);
    }

    // scores S[16 x 64] = Q(16x64) . K^T : 4 col-tiles x 2 K-steps = 8 WMMA
    float8 sc[4] = {};
    #pragma unroll
    for (int nt = 0; nt < 4; ++nt) {
      sc[nt] = wmma_f16(qa0, load_frag_b(Kp, HD, kb + nt * 16, 0,  lane), sc[nt]);
      sc[nt] = wmma_f16(qa1, load_frag_b(Kp, HD, kb + nt * 16, 32, lane), sc[nt]);
    }

    // scale + mask + one online-softmax pass over 64 keys
    #pragma unroll
    for (int r = 0; r < 8; ++r) {
      int qrow = qbase + r + 8 * hlf;
      const unsigned char* mq = mp + (size_t)qrow * SS + kb + nlane;
      #pragma unroll
      for (int nt = 0; nt < 4; ++nt)
        sc[nt][r] = mq[nt * 16] ? sc[nt][r] * SCALE : -1e9f;

      float mx = fmaxf(fmaxf(sc[0][r], sc[1][r]), fmaxf(sc[2][r], sc[3][r]));
      #pragma unroll
      for (int off = 8; off >= 1; off >>= 1)
        mx = fmaxf(mx, __shfl_xor(mx, off, 16));
      float mnew  = fmaxf(mrow[r], mx);
      float alpha = __expf(mrow[r] - mnew);
      mrow[r] = mnew;
      float rs = 0.0f;
      #pragma unroll
      for (int nt = 0; nt < 4; ++nt) {
        sc[nt][r] = __expf(sc[nt][r] - mnew);
        rs += sc[nt][r];
      }
      #pragma unroll
      for (int off = 8; off >= 1; off >>= 1)
        rs += __shfl_xor(rs, off, 16);
      lrow[r] = lrow[r] * alpha + rs;
      #pragma unroll
      for (int c = 0; c < 4; ++c) o[c][r] *= alpha;

      // stash P (C-layout -> row-major [16][64] LDS tile) for A-fragment reload
      int prow = r + 8 * hlf;
      #pragma unroll
      for (int nt = 0; nt < 4; ++nt)
        plds[wv][prow * 64 + nt * 16 + nlane] = (_Float16)sc[nt][r];
    }

    // O(16x64) += P(16x64) . V(64x64): 4 col-tiles x 2 K-steps = 8 WMMA
    half16 pa0 = load_frag_a(&plds[wv][0], 64, 0, 0,  lane);
    half16 pa1 = load_frag_a(&plds[wv][0], 64, 0, 32, lane);
    #pragma unroll
    for (int c = 0; c < 4; ++c) {
      o[c] = wmma_f16(pa0, load_frag_b(Vp, SS, c * 16, kb,      lane), o[c]);
      o[c] = wmma_f16(pa1, load_frag_b(Vp, SS, c * 16, kb + 32, lane), o[c]);
    }
  }

  // epilogue: O stored concat-head row-major [B][S][512] as f16
  #pragma unroll
  for (int r = 0; r < 8; ++r) {
    int qrow  = qbase + r + 8 * hlf;
    float inv = 1.0f / lrow[r];
    _Float16* op = O + ((size_t)b * SS + qrow) * DD + h * HD + nlane;
    #pragma unroll
    for (int c = 0; c < 4; ++c)
      op[c * 16] = (_Float16)(o[c][r] * inv);
  }
}

// ---------------------------------------------------------------------------
// Kernel 3: output projection.  out = O(f16) @ W_o(f32) + b_o, f32 result.
// ---------------------------------------------------------------------------
__global__ __launch_bounds__(256) void mha_proj_out(const _Float16* __restrict__ A,
                                                    const float* __restrict__ W,
                                                    const float* __restrict__ bias,
                                                    float* __restrict__ out) {
  __shared__ __align__(16) _Float16 wt[64 * 32];
  const int lane    = threadIdx.x & 31;
  const int wv      = threadIdx.x >> 5;
  const int rowbase = blockIdx.x * 128 + wv * 16;
  const int n0      = blockIdx.y * 64;
  const int hlf     = lane >> 4;
  const int nlane   = lane & 15;

  float8 acc[4] = {};

  for (int kk = 0; kk < DD; kk += 32) {
    __syncthreads();
    #pragma unroll
    for (int i = 0; i < 8; ++i) {
      int e   = threadIdx.x + i * 256;
      int col = e & 63;
      int kr  = e >> 6;
      wt[col * 32 + kr] = (_Float16)W[(size_t)(kk + kr) * DD + n0 + col];
    }
    __syncthreads();

    half16 a = load_frag_a(A, DD, rowbase, kk, lane);
    #pragma unroll
    for (int c = 0; c < 4; ++c)
      acc[c] = wmma_f16(a, load_frag_b(wt, 32, c * 16, 0, lane), acc[c]);
  }

  #pragma unroll
  for (int c = 0; c < 4; ++c) {
    #pragma unroll
    for (int r = 0; r < 8; ++r) {
      int grow = rowbase + r + 8 * hlf;
      int col  = n0 + c * 16 + nlane;
      out[(size_t)grow * DD + col] = acc[c][r] + bias[col];
    }
  }
}

// ---------------------------------------------------------------------------
extern "C" void kernel_launch(void* const* d_in, const int* in_sizes, int n_in,
                              void* d_out, int out_size, void* d_ws, size_t ws_size,
                              hipStream_t stream) {
  const float*         x_q  = (const float*)d_in[0];
  const float*         x_k  = (const float*)d_in[1];
  const float*         x_v  = (const float*)d_in[2];
  const unsigned char* mask = (const unsigned char*)d_in[3];
  const float*         W_q  = (const float*)d_in[4];
  const float*         b_q  = (const float*)d_in[5];
  const float*         W_k  = (const float*)d_in[6];
  const float*         b_k  = (const float*)d_in[7];
  const float*         W_v  = (const float*)d_in[8];
  const float*         b_v  = (const float*)d_in[9];
  const float*         W_o  = (const float*)d_in[10];
  const float*         b_o  = (const float*)d_in[11];

  // workspace: Q, K, V^T, O  — each B*H*S*hd = 4,194,304 f16 (8 MB), 32 MB total
  const size_t qkv = (size_t)BB * HH * SS * HD;
  _Float16* Qws = (_Float16*)d_ws;
  _Float16* Kws = Qws + qkv;
  _Float16* Vws = Qws + 2 * qkv;
  _Float16* Ows = Qws + 3 * qkv;

  dim3 gproj(8192 / 128, DD / 64);      // 64 x 8 blocks
  dim3 blk(256);

  mha_proj_qkv<<<gproj, blk, 0, stream>>>(x_q, W_q, b_q, Qws, 0);
  mha_proj_qkv<<<gproj, blk, 0, stream>>>(x_k, W_k, b_k, Kws, 0);
  mha_proj_qkv<<<gproj, blk, 0, stream>>>(x_v, W_v, b_v, Vws, 1);

  dim3 gattn(SS / 128, HH, BB);         // 32 x 8 x 2 blocks
  mha_attn<<<gattn, blk, 0, stream>>>(Qws, Kws, Vws, mask, Ows);

  mha_proj_out<<<gproj, blk, 0, stream>>>(Ows, W_o, b_o, (float*)d_out);
}
